// GCN_LTFGW_parallel_82248623718954
// MI455X (gfx1250) — matrix operand order repaired and checked by reading, not compiled, not run on hardware
//
#include <hip/hip_runtime.h>
#include <math.h>

#define N_NODES 10000
#define DEG     16
#define HIDDEN  128
#define NTPL    16
#define NTN     8     // template nodes
#define M_SUB   17    // star subgraph nodes (center + DEG)
#define NCLS    16
#define FW_ITERS 4
#define SK_ITERS 10
#define SK_REG   0.05f

typedef __attribute__((ext_vector_type(16))) __bf16       v16bf;
typedef __attribute__((ext_vector_type(8)))  float        v8f;
typedef __attribute__((ext_vector_type(4)))  unsigned int u32x4;
typedef __attribute__((ext_vector_type(4)))  float        f32x4;

union Frag { v16bf v; u32x4 q[2]; };

__device__ __forceinline__ unsigned short f2bf(float f) {
  unsigned int u = __float_as_uint(f);
  unsigned int r = (u + 0x7FFFu + ((u >> 16) & 1u)) >> 16;  // RNE
  return (unsigned short)r;
}

// ---------------------------------------------------------------- prep ----
__global__ void k_prep_x(const float* __restrict__ x,
                         unsigned short* __restrict__ xbf, int n) {
  int i = blockIdx.x * blockDim.x + threadIdx.x;
  if (i < n) xbf[i] = f2bf(x[i]);
}

__global__ void k_prep_wt(const float* __restrict__ W1, const float* __restrict__ W2,
                          const float* __restrict__ tF, const float* __restrict__ tC,
                          unsigned short* __restrict__ W1T, unsigned short* __restrict__ W2T,
                          unsigned short* __restrict__ tFb,
                          float* __restrict__ fnorm, float* __restrict__ tcq) {
  int i = blockIdx.x * blockDim.x + threadIdx.x;   // 16384 threads
  if (i < 128 * 128) {
    int n = i >> 7, k = i & 127;
    W1T[n * 128 + k] = f2bf(W1[k * 128 + n]);      // B^T so GEMM B loads contiguous
    W2T[n * 128 + k] = f2bf(W2[k * 128 + n]);
    tFb[i] = f2bf(tF[i]);                          // [tj][d] layout already B^T
  }
  if (i < 128) {                                   // ||F_tj||^2
    float s = 0.f;
    for (int d = 0; d < 128; ++d) { float v = tF[i * 128 + d]; s += v * v; }
    fnorm[i] = s;
  }
  if (i < NTPL * NTN) {                            // (tC^2 @ q)[t,j]
    float s = 0.f;
    for (int l = 0; l < NTN; ++l) { float v = tC[i * NTN + l]; s += v * v; }
    tcq[i] = s * (1.0f / NTN);
  }
}

// ---------------------------------------------------- WMMA GEMM (x @ W) ----
// C[10000,128] = A[10000,128] @ B[128,128]; BT = B^T bf16. One 16x16 tile/wave.
__global__ void __launch_bounds__(256) k_gemm(const unsigned short* __restrict__ A,
                                              const unsigned short* __restrict__ BT,
                                              float* __restrict__ C) {
  const int mt = blockIdx.x;            // 625 M-tiles (10000 = 625*16 exactly)
  const int l  = threadIdx.x & 31;
  const int nt = threadIdx.x >> 5;      // wave id == N-tile (8 waves, N=128)
  const int row   = mt * 16 + (l & 15);
  const int khalf = (l >> 4) * 8;       // A: K sub-halves per ISA layout
  const int n     = nt * 16 + (l & 15);
  const int kb16  = (l >> 4) * 16;      // unused: B loads use khalf, same k order as A
  v8f acc = {};
  #pragma unroll
  for (int kk = 0; kk < 4; ++kk) {      // K = 128 = 4 * 32
    Frag a, b;
    const unsigned short* ap = A + row * 128 + kk * 32;
    a.q[0] = *(const u32x4*)(ap + khalf);
    a.q[1] = *(const u32x4*)(ap + 16 + khalf);
    const unsigned short* bp = BT + n * 128 + kk * 32 + khalf;
    b.q[0] = *(const u32x4*)(bp);
    b.q[1] = *(const u32x4*)(bp + 16);
    acc = __builtin_amdgcn_wmma_f32_16x16x32_bf16(false, a.v, false, b.v,
                                                  (short)0, acc, false, false);
  }
  const int rbase = mt * 16 + ((l >> 4) << 3);   // C: lanes 16-31 hold M+8
  #pragma unroll
  for (int r = 0; r < 8; ++r) C[(rbase + r) * 128 + n] = acc[r];
}

// ----------------------------------- GCN aggregate (deg == 17 uniformly) ----
__global__ void k_aggregate(const float* __restrict__ XW1, const float* __restrict__ XW2,
                            const int* __restrict__ src,
                            const float* __restrict__ b1, const float* __restrict__ b2,
                            float* __restrict__ h1, float* __restrict__ h2) {
  const int i = blockIdx.x, t = threadIdx.x;     // 10000 blocks x 128 threads
  float s1 = XW1[i * 128 + t], s2 = XW2[i * 128 + t];
  for (int j = 0; j < DEG; ++j) {
    int nb = src[i * DEG + j];
    s1 += XW1[nb * 128 + t];
    s2 += XW2[nb * 128 + t];
  }
  const float inv = 1.0f / 17.0f;
  float v1 = s1 * inv + b1[t];
  float v2 = s2 * inv + b2[t];
  h1[i * 128 + t] = v1 > 0.f ? v1 : 0.f;
  h2[i * 128 + t] = v2 > 0.f ? v2 : 0.f;
}

// -------------------------------------------- fused per-node LTFGW block ----
// 1 block / node. 8 waves; wave w computes Mcost columns for templates 2w,2w+1
// via WMMA, then runs FW+Sinkhorn for those same templates (wave-private LDS,
// no barriers needed: same-wave LDS ops are in order).
__global__ void __launch_bounds__(256) k_ltfgw(
    const float* __restrict__ h1, const int* __restrict__ src,
    const unsigned short* __restrict__ tFb,
    const float* __restrict__ fnorm, const float* __restrict__ tcq,
    const float* __restrict__ tC, const float* __restrict__ alpha0,
    float* __restrict__ y) {
  __shared__ __attribute__((aligned(16))) unsigned short Xb[32 * 128]; // bf16, rows 17..31 zero
  __shared__ float xnorm[32];
  __shared__ float Mc[NTPL][M_SUB * NTN];     // feature cost [t][k*8+j]
  __shared__ float red[NTPL][16][NTN];        // per-template cross-lane scratch
  __shared__ float row0buf[NTPL][NTN];

  const int node = blockIdx.x;
  const int tid  = threadIdx.x;
  const int w    = tid >> 5;
  const int l    = tid & 31;

  // phase 0: zero pad rows of X tile
  {
    unsigned int* p = (unsigned int*)Xb;      // 2048 dwords
    for (int i = tid; i < 2048; i += 256) p[i] = 0u;
  }
  __syncthreads();

  // phase 1: stage rows 0..16 (center + neighbors) as bf16, compute ||X_k||^2
  for (int rr = 0; rr < 3; ++rr) {
    int row = w + rr * 8;                     // wave-uniform
    if (row <= 16) {
      int srow = (row == 0) ? node : src[node * DEG + (row - 1)];
      f32x4 v4 = *(const f32x4*)(h1 + srow * 128 + l * 4);
      unsigned short* d = Xb + row * 128 + l * 4;
      d[0] = f2bf(v4[0]); d[1] = f2bf(v4[1]); d[2] = f2bf(v4[2]); d[3] = f2bf(v4[3]);
      float s = v4[0]*v4[0] + v4[1]*v4[1] + v4[2]*v4[2] + v4[3]*v4[3];
      for (int off = 16; off > 0; off >>= 1) s += __shfl_down(s, off);
      if (l == 0) xnorm[row] = s;
    }
  }
  __syncthreads();

  // phase 2: WMMA  X[32,128] @ tF^T[128,128] -> Mcost columns for this wave
  const int n     = w * 16 + (l & 15);        // global tj column
  const int khalf = (l >> 4) * 8;
  const int kb16  = (l >> 4) * 16;
  const float fn  = fnorm[n];
  #pragma unroll
  for (int mt = 0; mt < 2; ++mt) {
    v8f acc = {};
    const int arow = mt * 16 + (l & 15);
    #pragma unroll
    for (int kk = 0; kk < 4; ++kk) {
      Frag a, b;
      const unsigned short* ap = Xb + arow * 128 + kk * 32;
      a.q[0] = *(const u32x4*)(ap + khalf);           // ds_load_b128
      a.q[1] = *(const u32x4*)(ap + 16 + khalf);
      const unsigned short* bp = tFb + n * 128 + kk * 32 + khalf;
      b.q[0] = *(const u32x4*)(bp);
      b.q[1] = *(const u32x4*)(bp + 16);
      acc = __builtin_amdgcn_wmma_f32_16x16x32_bf16(false, a.v, false, b.v,
                                                    (short)0, acc, false, false);
    }
    const int rbase = mt * 16 + ((l >> 4) << 3);
    const int t0 = n >> 3, j0 = n & 7;
    #pragma unroll
    for (int r = 0; r < 8; ++r) {
      int k = rbase + r;
      if (k < M_SUB) Mc[t0][k * NTN + j0] = xnorm[k] + fn - 2.0f * acc[r];
    }
  }
  // (no barrier: this wave wrote exactly the Mc[t] slices it consumes below)

  // phase 3: FW + Sinkhorn. 16 lanes per template; lane owns row l16 (+row 16
  // on lane 0). Star graph => C1@T has only 2 distinct rows (B0 for k=0, BR
  // for k>=1).
  const int t   = 2 * w + (l >> 4);
  const int l16 = l & 15;
  const int rc  = (l16 == 0) ? 2 : 1;
  int rowk[2]; rowk[0] = l16; rowk[1] = 16;

  float Mrow[2][NTN], Trow[2][NTN], Krow[2][NTN];
  for (int r = 0; r < rc; ++r)
    for (int j = 0; j < NTN; ++j) Mrow[r][j] = Mc[t][rowk[r] * NTN + j];
  float tcqv[NTN];
  for (int j = 0; j < NTN; ++j) tcqv[j] = tcq[t * NTN + j];
  const float tinit = 1.0f / (17.0f * 8.0f);
  for (int r = 0; r < 2; ++r)
    for (int j = 0; j < NTN; ++j) Trow[r][j] = tinit;

  const float logp  = -logf(17.0f);
  const float logq  = -logf(8.0f);
  const float a0    = alpha0[0];
  const float alpha = 1.0f / (1.0f + __expf(-a0));

  auto computeB = [&](float B0[NTN], float BR[NTN]) {
    // colsum over rows k>=1 (cross-lane) and broadcast of row 0
    for (int j = 0; j < NTN; ++j)
      red[t][l16][j] = (l16 >= 1) ? Trow[0][j] : Trow[1][j];   // lane0 adds row16
    if (l16 == 0)
      for (int j = 0; j < NTN; ++j) row0buf[t][j] = Trow[0][j];
    float colsum[NTN], row0v[NTN];
    for (int j = 0; j < NTN; ++j) {
      float s = 0.f;
      for (int ll = 0; ll < 16; ++ll) s += red[t][ll][j];
      colsum[j] = s;
      row0v[j]  = row0buf[t][j];
    }
    const float* tCt = tC + t * 64;           // symmetric
    for (int j = 0; j < NTN; ++j) {
      float s0 = 0.f, sr = 0.f;
      for (int jj = 0; jj < NTN; ++jj) {
        float c = tCt[j * NTN + jj];
        s0 += colsum[jj] * c;
        sr += row0v[jj] * c;
      }
      B0[j] = s0; BR[j] = sr;
    }
  };

  float B0[NTN], BR[NTN];
  for (int it = 0; it < FW_ITERS; ++it) {
    computeB(B0, BR);
    // G = (1-a)M + 2a*(constC - 2B); per-template absmax normalize -> logK
    float amax = 0.f;
    for (int r = 0; r < rc; ++r) {
      int k = rowk[r];
      float c1p = (k == 0) ? (16.0f / 17.0f) : (1.0f / 17.0f);
      const float* Bv = (k == 0) ? B0 : BR;
      for (int j = 0; j < NTN; ++j) {
        float g = (1.0f - alpha) * Mrow[r][j] +
                  2.0f * alpha * ((c1p + tcqv[j]) - 2.0f * Bv[j]);
        Krow[r][j] = g;
        float ag = fabsf(g); amax = ag > amax ? ag : amax;
      }
    }
    red[t][l16][0] = amax;
    for (int ll = 0; ll < 16; ++ll) { float v = red[t][ll][0]; amax = v > amax ? v : amax; }
    float ksc = -1.0f / ((amax + 1e-9f) * SK_REG);
    for (int r = 0; r < rc; ++r)
      for (int j = 0; j < NTN; ++j) Krow[r][j] *= ksc;

    float gv[NTN]; for (int j = 0; j < NTN; ++j) gv[j] = 0.f;
    float fv[2] = {0.f, 0.f};
    for (int si = 0; si < SK_ITERS; ++si) {
      for (int r = 0; r < rc; ++r) {          // f = logp - LSE_j
        float m = -1e30f;
        for (int j = 0; j < NTN; ++j) { float v = Krow[r][j] + gv[j]; m = v > m ? v : m; }
        float s = 0.f;
        for (int j = 0; j < NTN; ++j) s += __expf(Krow[r][j] + gv[j] - m);
        fv[r] = logp - (m + __logf(s));
      }
      float Mj[NTN];                          // g = logq - LSE_k (cross-lane)
      for (int j = 0; j < NTN; ++j) {
        float m = Krow[0][j] + fv[0];
        if (rc == 2) { float v = Krow[1][j] + fv[1]; m = v > m ? v : m; }
        red[t][l16][j] = m;
      }
      for (int j = 0; j < NTN; ++j) {
        float m = -1e30f;
        for (int ll = 0; ll < 16; ++ll) { float v = red[t][ll][j]; m = v > m ? v : m; }
        Mj[j] = m;
      }
      for (int j = 0; j < NTN; ++j) {
        float s = __expf(Krow[0][j] + fv[0] - Mj[j]);
        if (rc == 2) s += __expf(Krow[1][j] + fv[1] - Mj[j]);
        red[t][l16][j] = s;
      }
      for (int j = 0; j < NTN; ++j) {
        float s = 0.f;
        for (int ll = 0; ll < 16; ++ll) s += red[t][ll][j];
        gv[j] = logq - (Mj[j] + __logf(s));
      }
    }
    float step = 2.0f / (float)(it + 3);
    for (int r = 0; r < rc; ++r)
      for (int j = 0; j < NTN; ++j) {
        float tn = __expf(fv[r] + Krow[r][j] + gv[j]);
        Trow[r][j] = (1.0f - step) * Trow[r][j] + step * tn;
      }
  }

  // final FGW distance for this template
  computeB(B0, BR);
  float local = 0.f;
  for (int r = 0; r < rc; ++r) {
    int k = rowk[r];
    float c1p = (k == 0) ? (16.0f / 17.0f) : (1.0f / 17.0f);
    const float* Bv = (k == 0) ? B0 : BR;
    for (int j = 0; j < NTN; ++j) {
      float tens = (c1p + tcqv[j]) - 2.0f * Bv[j];
      local += ((1.0f - alpha) * Mrow[r][j] + alpha * tens) * Trow[r][j];
    }
  }
  red[t][l16][0] = local;
  if (l16 == 0) {
    float s = 0.f;
    for (int ll = 0; ll < 16; ++ll) s += red[t][ll][0];
    y[node * NTPL + t] = s;
  }
}

// ------------------------------------------------------- batchnorm stats ----
__global__ void k_bnstats(const float* __restrict__ h2, const float* __restrict__ y,
                          float* __restrict__ mu, float* __restrict__ var) {
  __shared__ float ssum[256], ssq[256];
  const int c = blockIdx.x;                   // 144 columns
  float s = 0.f, sq = 0.f;
  for (int i = threadIdx.x; i < N_NODES; i += 256) {
    float v = (c < HIDDEN) ? h2[i * HIDDEN + c] : y[i * NTPL + (c - HIDDEN)];
    s += v; sq += v * v;
  }
  ssum[threadIdx.x] = s; ssq[threadIdx.x] = sq;
  __syncthreads();
  for (int off = 128; off > 0; off >>= 1) {
    if (threadIdx.x < off) {
      ssum[threadIdx.x] += ssum[threadIdx.x + off];
      ssq[threadIdx.x]  += ssq[threadIdx.x + off];
    }
    __syncthreads();
  }
  if (threadIdx.x == 0) {
    float m = ssum[0] / (float)N_NODES;
    mu[c]  = m;
    var[c] = ssq[0] / (float)N_NODES - m * m;
  }
}

// ----------------------------------------------------- BN + final linear ----
__global__ void k_final(const float* __restrict__ h2, const float* __restrict__ y,
                        const float* __restrict__ mu, const float* __restrict__ var,
                        const float* __restrict__ gamma, const float* __restrict__ beta,
                        const float* __restrict__ linW, const float* __restrict__ linb,
                        float* __restrict__ out) {
  int idx = blockIdx.x * blockDim.x + threadIdx.x;
  if (idx >= N_NODES * NCLS) return;
  int i = idx / NCLS, c = idx % NCLS;
  float acc = linb[c];
  for (int f = 0; f < HIDDEN + NTPL; ++f) {
    float v  = (f < HIDDEN) ? h2[i * HIDDEN + f] : y[i * NTPL + (f - HIDDEN)];
    float zh = (v - mu[f]) * rsqrtf(var[f] + 1e-5f) * gamma[f] + beta[f];
    acc += zh * linW[f * NCLS + c];
  }
  out[idx] = acc;
}

// --------------------------------------------------------------- launch ----
extern "C" void kernel_launch(void* const* d_in, const int* in_sizes, int n_in,
                              void* d_out, int out_size, void* d_ws, size_t ws_size,
                              hipStream_t stream) {
  (void)in_sizes; (void)n_in; (void)out_size; (void)ws_size;
  const float* x    = (const float*)d_in[0];
  const int*   ei   = (const int*)d_in[1];
  const float* W1   = (const float*)d_in[2];
  const float* b1   = (const float*)d_in[3];
  const float* W2   = (const float*)d_in[4];
  const float* b2   = (const float*)d_in[5];
  const float* tC   = (const float*)d_in[6];
  const float* tF   = (const float*)d_in[7];
  const float* al0  = (const float*)d_in[8];
  const float* gam  = (const float*)d_in[9];
  const float* bet  = (const float*)d_in[10];
  const float* linW = (const float*)d_in[11];
  const float* linb = (const float*)d_in[12];
  float* out = (float*)d_out;
  const int* src = ei;                        // edge_index[0]

  char* p = (char*)d_ws;
  auto carve = [&](size_t bytes) -> char* {
    char* r = p; p += (bytes + 255) & ~(size_t)255; return r;
  };
  unsigned short* xbf = (unsigned short*)carve((size_t)N_NODES * 128 * 2);
  unsigned short* W1T = (unsigned short*)carve(128 * 128 * 2);
  unsigned short* W2T = (unsigned short*)carve(128 * 128 * 2);
  unsigned short* tFb = (unsigned short*)carve(128 * 128 * 2);
  float* fnorm = (float*)carve(128 * 4);
  float* tcq   = (float*)carve(128 * 4);
  float* XW1   = (float*)carve((size_t)N_NODES * 128 * 4);
  float* XW2   = (float*)carve((size_t)N_NODES * 128 * 4);
  float* h1    = (float*)carve((size_t)N_NODES * 128 * 4);
  float* h2    = (float*)carve((size_t)N_NODES * 128 * 4);
  float* yb    = (float*)carve((size_t)N_NODES * NTPL * 4);
  float* mu    = (float*)carve(144 * 4);
  float* var   = (float*)carve(144 * 4);

  k_prep_x<<<(N_NODES * 128 + 255) / 256, 256, 0, stream>>>(x, xbf, N_NODES * 128);
  k_prep_wt<<<64, 256, 0, stream>>>(W1, W2, tF, tC, W1T, W2T, tFb, fnorm, tcq);
  k_gemm<<<625, 256, 0, stream>>>(xbf, W1T, XW1);
  k_gemm<<<625, 256, 0, stream>>>(xbf, W2T, XW2);
  k_aggregate<<<N_NODES, 128, 0, stream>>>(XW1, XW2, src, b1, b2, h1, h2);
  k_ltfgw<<<N_NODES, 256, 0, stream>>>(h1, src, tFb, fnorm, tcq, tC, al0, yb);
  k_bnstats<<<144, 256, 0, stream>>>(h2, yb, mu, var);
  k_final<<<(N_NODES * NCLS + 255) / 256, 256, 0, stream>>>(
      h2, yb, mu, var, gam, bet, linW, linb, out);
}
